// NewMultiHeadAttention_81930796138921
// MI455X (gfx1250) — matrix operand order repaired and checked by reading, not compiled, not run on hardware
//
#include <hip/hip_runtime.h>

// ---------------------------------------------------------------------------
// PaTH attention for MI455X (gfx1250, wave32, WMMA).
// All matmul-shaped work uses v_wmma_f32_16x16x32_bf16 with f32 accumulation.
// ---------------------------------------------------------------------------

typedef __attribute__((ext_vector_type(16))) __bf16 v16bf;
typedef __attribute__((ext_vector_type(8)))  __bf16 v8bf;
typedef __attribute__((ext_vector_type(8)))  float  v8f;

constexpr int NB  = 2;
constexpr int NT  = 512;
constexpr int ND  = 1024;
constexpr int NH  = 16;
constexpr int NDH = 64;
constexpr int NR  = 32;
constexpr int NBT = NB * NT;   // 1024
constexpr int NBH = NB * NH;   // 32
constexpr int TP  = 544;       // padded stride for v^T (P-tile K=32 overrun)

// ---- WMMA fragment helpers (layouts per CDNA5 ISA 7.12.2, wave32) ----------

// A-operand 16x32 bf16 from row-major [rows][ld]: lane m=L&15 holds row m,
// K-chunks [8h,8h+8) and [16+8h,16+8h+8) -> two contiguous 16B loads.
__device__ __forceinline__ v16bf load_a_frag(const __bf16* base, int ld) {
  const int lane = threadIdx.x & 31;
  const int m = lane & 15, hf = lane >> 4;
  const __bf16* p = base + m * ld + 8 * hf;
  v8bf lo = *(const v8bf*)p;
  v8bf hi = *(const v8bf*)(p + 16);
  v16bf r;
#pragma unroll
  for (int i = 0; i < 8; ++i) { r[i] = lo[i]; r[i + 8] = hi[i]; }
  return r;
}

// B-operand 32x16 bf16 from TRANSPOSED storage Bt[n][k] row-major: lane
// n=L&15 reads Bt[n][8h .. 8h+8) and Bt[n][16+8h .. 16+8h+8) -> two 16B loads.
__device__ __forceinline__ v16bf load_b_frag(const __bf16* baseT, int ld) {
  const int lane = threadIdx.x & 31;
  const int n = lane & 15, hf = lane >> 4;
  const __bf16* p = baseT + n * ld + 8 * hf;
  v8bf lo = *(const v8bf*)p;
  v8bf hi = *(const v8bf*)(p + 16);
  v16bf r;
#pragma unroll
  for (int i = 0; i < 8; ++i) { r[i] = lo[i]; r[i + 8] = hi[i]; }
  return r;
}

__device__ __forceinline__ v8f wmma_bf16(v16bf a, v16bf b, v8f c) {
  // (neg_a, A, neg_b, B, c_mod, C, reuse_a, reuse_b)
  return __builtin_amdgcn_wmma_f32_16x16x32_bf16(false, a, false, b,
                                                 (short)0, c, false, false);
}

// DS ops are in-order per wave; this fences the compiler and waits the DScnt
// so LDS-staged data is valid before re-reading it in a different layout.
__device__ __forceinline__ void lds_fence() {
  asm volatile("s_wait_dscnt 0" ::: "memory");
}

// ---- elementwise / conversion kernels --------------------------------------

__global__ void cvt_bf16_kernel(const float* __restrict__ in,
                                __bf16* __restrict__ out, int n) {
  int i = blockIdx.x * 256 + threadIdx.x;
  if (i < n) out[i] = (__bf16)in[i];
}

__global__ void transpose_cvt_kernel(const float* __restrict__ W,
                                     __bf16* __restrict__ Wt, int K, int N) {
  int i = blockIdx.x * 256 + threadIdx.x;
  if (i < K * N) {
    int nn = i / K, kk = i % K;
    Wt[i] = (__bf16)W[(size_t)kk * N + nn];
  }
}

// ---- generic bf16 GEMM: C[M,N] = A[M,K] * Bt[N,K]^T ------------------------

__global__ void __launch_bounds__(256) gemm_bf16_kernel(
    const __bf16* __restrict__ A, const __bf16* __restrict__ Bt,
    float* __restrict__ C, int M, int N, int K) {
  const int wave = (blockIdx.x * blockDim.x + threadIdx.x) >> 5;
  const int tilesN = N >> 6;                  // 64-wide N strips per wave
  const int mi = wave / tilesN;
  const int nj = wave % tilesN;
  if (mi >= (M >> 4)) return;
  const int lane = threadIdx.x & 31;
  const int n = lane & 15, hf = lane >> 4;
  v8f acc[4] = {};
  for (int kk = 0; kk < K; kk += 32) {
    v16bf a = load_a_frag(A + (size_t)mi * 16 * K + kk, K);
#pragma unroll
    for (int t = 0; t < 4; ++t) {
      v16bf b = load_b_frag(Bt + (size_t)(nj * 64 + t * 16) * K + kk, K);
      acc[t] = wmma_bf16(a, b, acc[t]);
    }
  }
#pragma unroll
  for (int t = 0; t < 4; ++t)
#pragma unroll
    for (int r = 0; r < 8; ++r)
      C[(size_t)(mi * 16 + r + 8 * hf) * N + nj * 64 + t * 16 + n] = acc[t][r];
}

// ---- low-rank w projection, conv, gates ------------------------------------

__global__ void lowrank1_kernel(const float* __restrict__ x,
                                const float* __restrict__ wA,
                                float* __restrict__ P) {
  int i = blockIdx.x * 256 + threadIdx.x;     // BT*R
  if (i >= NBT * NR) return;
  int bt = i / NR, r = i % NR;
  float acc = 0.f;
  for (int dd = 0; dd < ND; ++dd)
    acc += x[(size_t)bt * ND + dd] * wA[(size_t)dd * NR + r];
  P[i] = acc;
}

__global__ void lowrank2_kernel(const float* __restrict__ P,
                                const float* __restrict__ wBm,
                                float* __restrict__ w0) {
  int i = blockIdx.x * 256 + threadIdx.x;     // BT*D
  if (i >= NBT * ND) return;
  int bt = i / ND, c = i % ND;
  float acc = 0.f;
  for (int r = 0; r < NR; ++r)
    acc += P[bt * NR + r] * wBm[(size_t)r * ND + c];
  w0[i] = acc;
}

__global__ void conv_silu_kernel(const float* __restrict__ w0,
                                 const float* __restrict__ cw,
                                 float* __restrict__ w1) {
  int i = blockIdx.x * 256 + threadIdx.x;     // BT*D
  if (i >= NBT * ND) return;
  int bt = i / ND, c = i % ND;
  int t = bt % NT;
  float acc = w0[i] * cw[c * 3 + 2];
  if (t >= 1) acc += w0[i - ND] * cw[c * 3 + 1];
  if (t >= 2) acc += w0[i - 2 * ND] * cw[c * 3 + 0];
  w1[i] = acc / (1.f + __expf(-acc));         // silu
}

__global__ void normpack_w_kernel(const float* __restrict__ w1,
                                  __bf16* __restrict__ wh) {
  int i = blockIdx.x * 256 + threadIdx.x;     // BT*H
  if (i >= NBT * NH) return;
  int bt = i / NH, h = i % NH;
  int b = bt / NT, t = bt % NT;
  const float* src = w1 + (size_t)bt * ND + h * NDH;
  float ss = 0.f;
  for (int j = 0; j < NDH; ++j) ss += src[j] * src[j];
  float rn = rsqrtf(ss + 1e-12f);
  __bf16* dst = wh + ((size_t)(b * NH + h) * NT + t) * NDH;
  for (int j = 0; j < NDH; ++j) dst[j] = (__bf16)(src[j] * rn);
}

__global__ void gates_kernel(const float* __restrict__ x,
                             const float* __restrict__ bw,
                             const float* __restrict__ gw,
                             const float* __restrict__ gb,
                             float* __restrict__ beta,
                             float* __restrict__ graw) {
  int i = blockIdx.x * 256 + threadIdx.x;     // BT*H
  if (i >= NBT * NH) return;
  int bt = i / NH, h = i % NH;
  int b = bt / NT, t = bt % NT;
  const float* xr = x + (size_t)bt * ND;
  float z1 = 0.f, z2 = 0.f;
  for (int dd = 0; dd < ND; ++dd) {
    float xv = xr[dd];
    z1 += xv * bw[dd * NH + h];
    z2 += xv * gw[dd * NH + h];
  }
  z2 += gb[h];
  int idx = (b * NH + h) * NT + t;
  beta[idx] = 2.f / (1.f + __expf(-z1));
  graw[idx] = (z2 <= 0.f) ? (z2 - log1pf(__expf(z2))) : (-log1pf(__expf(-z2)));
}

__global__ void cumsum_kernel(const float* __restrict__ graw,
                              float* __restrict__ G) {
  int bh = threadIdx.x;
  if (bh >= NBH) return;
  float run = 0.f;
  for (int t = 0; t < NT; ++t) { run += graw[bh * NT + t]; G[bh * NT + t] = run; }
}

__global__ void rmspack_kernel(const float* __restrict__ q,
                               const float* __restrict__ k,
                               const float* __restrict__ v,
                               const float* __restrict__ qnw,
                               const float* __restrict__ knw,
                               __bf16* __restrict__ qh, __bf16* __restrict__ kh,
                               __bf16* __restrict__ vt) {
  int i = blockIdx.x * 256 + threadIdx.x;     // BT*H
  if (i >= NBT * NH) return;
  int bt = i / NH, h = i % NH;
  int b = bt / NT, t = bt % NT;
  size_t src = (size_t)bt * ND + h * NDH;
  int bh = b * NH + h;
  float sq = 0.f, sk = 0.f;
  for (int j = 0; j < NDH; ++j) {
    float a = q[src + j]; sq += a * a;
    float c = k[src + j]; sk += c * c;
  }
  float rq = rsqrtf(sq * (1.f / NDH) + 1e-5f);
  float rk = rsqrtf(sk * (1.f / NDH) + 1e-5f);
  __bf16* qd = qh + ((size_t)bh * NT + t) * NDH;
  __bf16* kd = kh + ((size_t)bh * NT + t) * NDH;
  for (int j = 0; j < NDH; ++j) {
    qd[j] = (__bf16)(q[src + j] * rq * qnw[j]);
    kd[j] = (__bf16)(k[src + j] * rk * knw[j]);
    vt[((size_t)bh * NDH + j) * TP + t] = (__bf16)v[src + j];   // v^T per head
  }
}

// ---- blocked unit-lower-triangular solve, one wave per (b,h) ---------------
// Produces Atb[j][t] = bf16( beta_t * A[t][j] )  (transposed, beta folded),
// where (I + stril((w w^T) * beta_s)) A = stril(w k^T).
__global__ void __launch_bounds__(32) solve_kernel(
    const __bf16* wh, const __bf16* kh, const float* beta, __bf16* Atb) {
  __shared__ __attribute__((aligned(64))) __bf16 stg[16][32];
  __shared__ __attribute__((aligned(64))) float dW[16][16];
  __shared__ __attribute__((aligned(64))) float sol[16][16];
  __shared__ float bb[16];
  const int bh = blockIdx.x;
  const int lane = threadIdx.x;
  const int n = lane & 15, hf = lane >> 4;
  const __bf16* wB = wh + (size_t)bh * NT * NDH;
  const __bf16* kB = kh + (size_t)bh * NT * NDH;
  __bf16* AtB = Atb + (size_t)bh * NT * NT;
  const float* betaB = beta + bh * NT;

  for (int ti = 0; ti < 32; ++ti) {
    const int t0 = ti * 16;
    const v16bf wa0 = load_a_frag(wB + t0 * NDH, NDH);
    const v16bf wa1 = load_a_frag(wB + t0 * NDH + 32, NDH);
    // diagonal W-gram block (w_t . w_s), t,s in this block
    v8f dG = {};
    dG = wmma_bf16(wa0, load_b_frag(wB + t0 * NDH, NDH), dG);
    dG = wmma_bf16(wa1, load_b_frag(wB + t0 * NDH + 32, NDH), dG);
#pragma unroll
    for (int r = 0; r < 8; ++r) dW[r + 8 * hf][n] = dG[r];
    if (lane < 16) bb[lane] = betaB[t0 + lane];
    lds_fence();

    for (int jt = 0; jt <= ti; ++jt) {
      const int j0 = jt * 16;
      v8f acc = {};
      acc = wmma_bf16(wa0, load_b_frag(kB + j0 * NDH, NDH), acc);       // w.k^T
      acc = wmma_bf16(wa1, load_b_frag(kB + j0 * NDH + 32, NDH), acc);
      if (jt == ti) {                                  // strict lower (t>j)
#pragma unroll
        for (int r = 0; r < 8; ++r)
          if (r + 8 * hf <= n) acc[r] = 0.f;
      }
      // subtract sum_{s<t0} (w_t.w_s) * Ab[s][j]
      for (int sc = j0 & ~31; sc < t0; sc += 32) {
#pragma unroll
        for (int u = 0; u < 2; ++u) {
          const int scol = sc + 16 * u;
          if (scol < t0) {
            v8f g2 = {};
            g2 = wmma_bf16(wa0, load_b_frag(wB + scol * NDH, NDH), g2);
            g2 = wmma_bf16(wa1, load_b_frag(wB + scol * NDH + 32, NDH), g2);
#pragma unroll
            for (int r = 0; r < 8; ++r)
              stg[r + 8 * hf][n + 16 * u] = (__bf16)(-g2[r]);
          } else {
#pragma unroll
            for (int r = 0; r < 8; ++r)
              stg[r + 8 * hf][n + 16 * u] = (__bf16)0.f;
          }
        }
        lds_fence();
        const v16bf lA = load_a_frag(&stg[0][0], 32);
        const v16bf ab = load_b_frag(AtB + (size_t)j0 * NT + sc, NT);
        acc = wmma_bf16(lA, ab, acc);
        lds_fence();
      }
      // dump to LDS and do the serial 16x16 unit-diag solve (16 lanes)
#pragma unroll
      for (int r = 0; r < 8; ++r) sol[r + 8 * hf][n] = acc[r];
      lds_fence();
      if (lane < 16) {
        const int j = lane;
        for (int t = 0; t < 16; ++t) {
          float val = sol[t][j];
          for (int s = 0; s < t; ++s) val -= dW[t][s] * sol[s][j];
          if (jt == ti && t <= j) val = 0.f;
          sol[t][j] = bb[t] * val;                    // Ab = beta_t * A
        }
        v16bf outv;
#pragma unroll
        for (int t = 0; t < 16; ++t) outv[t] = (__bf16)sol[t][j];
        *(v16bf*)(AtB + (size_t)(j0 + j) * NT + t0) = outv;   // transposed
      }
      lds_fence();
    }
  }
}

// ---- flash-style scores/softmax/PV, one wave per (b,h,i-block) -------------
__global__ void __launch_bounds__(256) attn_kernel(
    const __bf16* __restrict__ qh, const __bf16* __restrict__ kh,
    const __bf16* __restrict__ wh, const __bf16* __restrict__ vt,
    const __bf16* Atb, const float* __restrict__ G,
    float* __restrict__ o) {
  __shared__ __attribute__((aligned(64))) __bf16 stg[8][16][32];
  const int wv = threadIdx.x >> 5;
  const int lane = threadIdx.x & 31;
  const int wg = blockIdx.x * 8 + wv;
  const int bh = wg >> 5;
  const int ib = wg & 31;
  const int b = bh >> 4, h = bh & 15;
  const int i0 = ib * 16;
  const int n = lane & 15, hf = lane >> 4;

  const __bf16* qB = qh + (size_t)bh * NT * NDH;
  const __bf16* kB = kh + (size_t)bh * NT * NDH;
  const __bf16* wB = wh + (size_t)bh * NT * NDH;
  const __bf16* vtB = vt + (size_t)bh * NDH * TP;
  const __bf16* AtB = Atb + (size_t)bh * NT * NT;
  const float* Gb = G + bh * NT;

  const v16bf qa0 = load_a_frag(qB + i0 * NDH, NDH);
  const v16bf qa1 = load_a_frag(qB + i0 * NDH + 32, NDH);

  float Gi[8], mrow[8], srow[8];
  v8f oacc[4] = {};
#pragma unroll
  for (int r = 0; r < 8; ++r) {
    Gi[r] = Gb[i0 + r + 8 * hf];
    mrow[r] = -3.0e38f;
    srow[r] = 0.f;
  }

  for (int jt = 0; jt <= ib; ++jt) {
    const int j0 = jt * 16;
    v8f acc = {};
    acc = wmma_bf16(qa0, load_b_frag(kB + j0 * NDH, NDH), acc);   // q.k^T
    acc = wmma_bf16(qa1, load_b_frag(kB + j0 * NDH + 32, NDH), acc);

    // S -= sum_t qw[i,t] * Ab[t,j]   (Ab already strictly lower + beta-folded)
    for (int tc = j0 & ~31; tc < i0 + 16; tc += 32) {
#pragma unroll
      for (int u = 0; u < 2; ++u) {
        const int tcol = tc + 16 * u;
        if (tcol < i0 + 16) {
          v8f dqw = {};
          dqw = wmma_bf16(qa0, load_b_frag(wB + tcol * NDH, NDH), dqw);
          dqw = wmma_bf16(qa1, load_b_frag(wB + tcol * NDH + 32, NDH), dqw);
#pragma unroll
          for (int r = 0; r < 8; ++r) {
            const int m = r + 8 * hf;
            float vq = dqw[r];
            if (tcol + n > i0 + m) vq = 0.f;          // qw masked to t <= i
            stg[wv][m][n + 16 * u] = (__bf16)(-vq);
          }
        } else {
#pragma unroll
          for (int r = 0; r < 8; ++r)
            stg[wv][r + 8 * hf][n + 16 * u] = (__bf16)0.f;
        }
      }
      lds_fence();
      const v16bf qwA = load_a_frag(&stg[wv][0][0], 32);
      const v16bf ab = load_b_frag(AtB + (size_t)j0 * NT + tc, NT);
      acc = wmma_bf16(qwA, ab, acc);
      lds_fence();
    }

    // logits + online softmax (row = M index; lanes of one 16-half share row)
    const float gj = Gb[j0 + n];
    float p[8];
#pragma unroll
    for (int r = 0; r < 8; ++r) {
      const int m = r + 8 * hf;
      float lg = acc[r] * 0.125f + Gi[r] - gj;        // scale = d^-0.5 = 1/8
      if (jt == ib && n > m) lg = -1.0e30f;           // causal mask
      float mx = lg;
      for (int msk = 1; msk < 16; msk <<= 1) mx = fmaxf(mx, __shfl_xor(mx, msk, 32));
      const float nm = fmaxf(mrow[r], mx);
      const float pe = __expf(lg - nm);
      float ps = pe;
      for (int msk = 1; msk < 16; msk <<= 1) ps += __shfl_xor(ps, msk, 32);
      const float corr = __expf(mrow[r] - nm);
      srow[r] = srow[r] * corr + ps;
      mrow[r] = nm;
      p[r] = pe;
#pragma unroll
      for (int t = 0; t < 4; ++t) oacc[t][r] *= corr;
    }

    // o += P * v  (stage P to A-layout, K padded to 32 with zeros)
#pragma unroll
    for (int r = 0; r < 8; ++r) {
      const int m = r + 8 * hf;
      stg[wv][m][n] = (__bf16)p[r];
      stg[wv][m][n + 16] = (__bf16)0.f;
    }
    lds_fence();
    const v16bf pA = load_a_frag(&stg[wv][0][0], 32);
#pragma unroll
    for (int t = 0; t < 4; ++t) {
      const v16bf vb = load_b_frag(vtB + (size_t)(t * 16) * TP + j0, TP);
      oacc[t] = wmma_bf16(pA, vb, oacc[t]);
    }
    lds_fence();
  }

#pragma unroll
  for (int t = 0; t < 4; ++t)
#pragma unroll
    for (int r = 0; r < 8; ++r) {
      const int m = r + 8 * hf;
      o[(size_t)(b * NT + i0 + m) * ND + h * NDH + t * 16 + n] =
          oacc[t][r] / srow[r];
    }
}

// ---------------------------------------------------------------------------

extern "C" void kernel_launch(void* const* d_in, const int* in_sizes, int n_in,
                              void* d_out, int out_size, void* d_ws,
                              size_t ws_size, hipStream_t stream) {
  (void)in_sizes; (void)n_in; (void)out_size; (void)ws_size;
  const float* x   = (const float*)d_in[0];
  const float* Wq  = (const float*)d_in[1];
  const float* Wk  = (const float*)d_in[2];
  const float* Wv  = (const float*)d_in[3];
  const float* Wo  = (const float*)d_in[4];
  const float* wA  = (const float*)d_in[5];
  const float* wBm = (const float*)d_in[6];
  const float* cw  = (const float*)d_in[7];
  const float* bw  = (const float*)d_in[8];
  const float* gw  = (const float*)d_in[9];
  const float* gb  = (const float*)d_in[10];
  const float* qnw = (const float*)d_in[11];
  const float* knw = (const float*)d_in[12];
  float* out = (float*)d_out;

  char* ws = (char*)d_ws;
  size_t cur = 0;
  auto alloc = [&](size_t bytes) {
    size_t off = cur;
    cur += (bytes + 255) & ~size_t(255);
    return off;
  };
  __bf16* xb   = (__bf16*)(ws + alloc((size_t)NBT * ND * 2));
  __bf16* Wqt  = (__bf16*)(ws + alloc((size_t)ND * ND * 2));
  __bf16* Wkt  = (__bf16*)(ws + alloc((size_t)ND * ND * 2));
  __bf16* Wvt  = (__bf16*)(ws + alloc((size_t)ND * ND * 2));
  __bf16* Wot  = (__bf16*)(ws + alloc((size_t)ND * ND * 2));
  float*  qf   = (float*)(ws + alloc((size_t)NBT * ND * 4));
  float*  kf   = (float*)(ws + alloc((size_t)NBT * ND * 4));
  float*  vf   = (float*)(ws + alloc((size_t)NBT * ND * 4));
  float*  Plr  = (float*)(ws + alloc((size_t)NBT * NR * 4));
  float*  w0   = (float*)(ws + alloc((size_t)NBT * ND * 4));
  float*  w1   = (float*)(ws + alloc((size_t)NBT * ND * 4));
  float*  betb = (float*)(ws + alloc((size_t)NBH * NT * 4));
  float*  grw  = (float*)(ws + alloc((size_t)NBH * NT * 4));
  float*  Gbuf = (float*)(ws + alloc((size_t)NBH * NT * 4));
  __bf16* qhb  = (__bf16*)(ws + alloc((size_t)NBH * NT * NDH * 2));
  __bf16* khb  = (__bf16*)(ws + alloc((size_t)NBH * NT * NDH * 2));
  __bf16* whb  = (__bf16*)(ws + alloc((size_t)NBH * NT * NDH * 2));
  __bf16* vtb  = (__bf16*)(ws + alloc((size_t)NBH * NDH * TP * 2));
  __bf16* Atbb = (__bf16*)(ws + alloc((size_t)NBH * NT * NT * 2));
  float*  obuf = (float*)(ws + alloc((size_t)NBT * ND * 4));
  __bf16* ob   = (__bf16*)(ws + alloc((size_t)NBT * ND * 2));

  const int nE = NBT * ND;                 // 1M elements
  cvt_bf16_kernel<<<nE / 256, 256, 0, stream>>>(x, xb, nE);
  transpose_cvt_kernel<<<(ND * ND) / 256, 256, 0, stream>>>(Wq, Wqt, ND, ND);
  transpose_cvt_kernel<<<(ND * ND) / 256, 256, 0, stream>>>(Wk, Wkt, ND, ND);
  transpose_cvt_kernel<<<(ND * ND) / 256, 256, 0, stream>>>(Wv, Wvt, ND, ND);
  transpose_cvt_kernel<<<(ND * ND) / 256, 256, 0, stream>>>(Wo, Wot, ND, ND);

  // QKV projections: 1024 waves = 128 blocks x 8 waves
  gemm_bf16_kernel<<<128, 256, 0, stream>>>(xb, Wqt, qf, NBT, ND, ND);
  gemm_bf16_kernel<<<128, 256, 0, stream>>>(xb, Wkt, kf, NBT, ND, ND);
  gemm_bf16_kernel<<<128, 256, 0, stream>>>(xb, Wvt, vf, NBT, ND, ND);

  lowrank1_kernel<<<(NBT * NR) / 256, 256, 0, stream>>>(x, wA, Plr);
  lowrank2_kernel<<<nE / 256, 256, 0, stream>>>(Plr, wBm, w0);
  conv_silu_kernel<<<nE / 256, 256, 0, stream>>>(w0, cw, w1);
  normpack_w_kernel<<<(NBT * NH) / 256, 256, 0, stream>>>(w1, whb);
  gates_kernel<<<(NBT * NH) / 256, 256, 0, stream>>>(x, bw, gw, gb, betb, grw);
  cumsum_kernel<<<1, 32, 0, stream>>>(grw, Gbuf);
  rmspack_kernel<<<(NBT * NH) / 256, 256, 0, stream>>>(qf, kf, vf, qnw, knw,
                                                       qhb, khb, vtb);

  hipMemsetAsync(Atbb, 0, (size_t)NBH * NT * NT * 2, stream);
  solve_kernel<<<NBH, 32, 0, stream>>>(whb, khb, betb, Atbb);
  attn_kernel<<<128, 256, 0, stream>>>(qhb, khb, whb, vtb, Atbb, Gbuf, obuf);

  cvt_bf16_kernel<<<nE / 256, 256, 0, stream>>>(obuf, ob, nE);
  gemm_bf16_kernel<<<128, 256, 0, stream>>>(ob, Wot, out, NBT, ND, ND);
}
